// DecoderLayer_14869176779448
// MI455X (gfx1250) — compile-verified, matrix operand order checked by reading, not run on hardware
//
#include <hip/hip_runtime.h>
#include <hip/hip_bf16.h>

// ---------------------------------------------------------------------------
// Types for WMMA fragments (gfx1250, wave32)
// ---------------------------------------------------------------------------
typedef __bf16 bf16_t;
typedef __attribute__((ext_vector_type(16))) bf16_t        v16bf;
typedef __attribute__((ext_vector_type(8)))  float         v8f;
typedef __attribute__((ext_vector_type(4)))  unsigned int  v4u;

union FragU {
    v16bf          b;
    v4u            q[2];
    unsigned short u[16];
};

__device__ __forceinline__ unsigned short f2bf(float f) {
    unsigned int x = __float_as_uint(f);
    unsigned int r = x + 0x7FFFu + ((x >> 16) & 1u);   // round-to-nearest-even
    return (unsigned short)(r >> 16);
}
__device__ __forceinline__ float bf2f(unsigned short h) {
    return __uint_as_float(((unsigned int)h) << 16);
}

// LDS byte offset of a __shared__ object (generic -> AS(3) -> 32-bit offset)
__device__ __forceinline__ unsigned int lds_off(const void* p) {
    return (unsigned int)(unsigned long long)(const __attribute__((address_space(3))) void*)p;
}

// Async global->LDS copy of 16 bytes (per-lane addresses), ASYNCcnt-tracked.
__device__ __forceinline__ void async_copy_b128(unsigned int lds_byte_addr,
                                                const void* gptr) {
    asm volatile("global_load_async_to_lds_b128 %0, %1, off"
                 :
                 : "v"(lds_byte_addr), "v"(gptr)
                 : "memory");
}
__device__ __forceinline__ void wait_async() {
    asm volatile("s_wait_asynccnt 0x0" ::: "memory");
}

// ---------------------------------------------------------------------------
// f32 -> bf16 convert (grid-stride)
// ---------------------------------------------------------------------------
__global__ __launch_bounds__(256) void cvt_f32_bf16(const float* __restrict__ x,
                                                    unsigned short* __restrict__ y,
                                                    long long n) {
    long long i = (long long)blockIdx.x * 256 + threadIdx.x;
    long long stride = (long long)gridDim.x * 256;
    for (; i < n; i += stride) y[i] = f2bf(x[i]);
}

// ---------------------------------------------------------------------------
// LayerNorm over D=1024, output bf16. One block (256 threads) per row.
// ---------------------------------------------------------------------------
__global__ __launch_bounds__(256) void ln_bf16(const float* __restrict__ x,
                                               const float* __restrict__ g,
                                               const float* __restrict__ bta,
                                               unsigned short* __restrict__ y,
                                               int rows) {
    __shared__ float s0[256];
    __shared__ float s1[256];
    int row = blockIdx.x;
    if (row >= rows) return;
    const float* xr = x + (size_t)row * 1024;
    float v[4];
    float a = 0.f, b = 0.f;
    for (int j = 0; j < 4; ++j) {
        v[j] = xr[threadIdx.x + 256 * j];
        a += v[j];
        b += v[j] * v[j];
    }
    s0[threadIdx.x] = a;
    s1[threadIdx.x] = b;
    __syncthreads();
    for (int o = 128; o > 0; o >>= 1) {
        if ((int)threadIdx.x < o) {
            s0[threadIdx.x] += s0[threadIdx.x + o];
            s1[threadIdx.x] += s1[threadIdx.x + o];
        }
        __syncthreads();
    }
    float m   = s0[0] * (1.f / 1024.f);
    float var = s1[0] * (1.f / 1024.f) - m * m;
    float inv = rsqrtf(var + 1e-5f);
    for (int j = 0; j < 4; ++j) {
        int c = threadIdx.x + 256 * j;
        y[(size_t)row * 1024 + c] = f2bf((v[j] - m) * inv * g[c] + bta[c]);
    }
}

// ---------------------------------------------------------------------------
// GEMM: out[M,N] = A[M,K](bf16) x Bw[K,N](bf16) (+bias)(+res)(ReLU?)
// Block: 256 threads = 8 waves; block tile 64(M) x 128(N); K-step 32.
// Wave tile: 16(M) x 64(N) = 4 WMMA accumulators reusing one A fragment.
// A tile copied global->LDS with GLOBAL_LOAD_ASYNC_TO_LDS_B128 (ASYNCcnt);
// B tile stored transposed [n][k] as packed 32-bit k-pairs (ds_store_b32).
// ---------------------------------------------------------------------------
__global__ __launch_bounds__(256) void gemm_bf16_wmma(
    const unsigned short* __restrict__ A, const unsigned short* __restrict__ Bw,
    const float* __restrict__ bias, const float* __restrict__ res,
    float* __restrict__ outF, unsigned short* __restrict__ outH,
    int M, int N, int K, int relu) {
    __shared__ __align__(16) unsigned short As[64 * 32];    // [m][k]   4KB
    __shared__ __align__(16) unsigned short Bs[128 * 32];   // [n][k]   8KB
    unsigned int* Bs32 = (unsigned int*)Bs;

    const int tid  = threadIdx.x;
    const int lane = tid & 31;
    const int wave = tid >> 5;
    const int lo = lane & 15, hi = lane >> 4;
    const int bn = blockIdx.x * 128;
    const int bm = blockIdx.y * 64;
    const int wmRow = (wave >> 1) * 16;   // 0,16,32,48 within block
    const int wnCol = (wave & 1) * 64;    // 0,64 within block

    v8f acc[4];
    for (int c = 0; c < 4; ++c)
        for (int i = 0; i < 8; ++i) acc[c][i] = 0.f;

    // A-tile loader mapping: one b128 per thread
    const int ar  = tid >> 2;            // 0..63
    const int ac  = (tid & 3) * 8;       // 0,8,16,24
    int arow = bm + ar; if (arow >= M) arow = M - 1;
    const unsigned int a_lds = lds_off(As + ar * 32 + ac);

    // B-tile loader mapping: k-pair p, 8 columns per thread
    const int bp_ = tid & 15;            // k-pair 0..15 -> k = 2p, 2p+1
    const int n8  = (tid >> 4) * 8;      // 0..120

    for (int k0 = 0; k0 < K; k0 += 32) {
        // ---- A tile: async global -> LDS (no transform needed) ----
        async_copy_b128(a_lds, A + (size_t)arow * K + k0 + ac);

        // ---- B tile: load 2 k-rows x 8 cols, pack k-pairs, transpose ----
        {
            const unsigned short* s0 = Bw + (size_t)(k0 + 2 * bp_) * N + bn + n8;
            union { v4u q; unsigned short h[8]; } r0, r1;
            r0.q = *(const v4u*)s0;
            r1.q = *(const v4u*)(s0 + N);
            for (int j = 0; j < 8; ++j)
                Bs32[(n8 + j) * 16 + bp_] =
                    (unsigned int)r0.h[j] | ((unsigned int)r1.h[j] << 16);
        }

        // prefetch next K tiles while this one is in flight
        if (k0 + 32 < K) {
            __builtin_prefetch(A + (size_t)arow * K + k0 + 32 + ac, 0, 1);
            __builtin_prefetch(Bw + (size_t)(k0 + 32 + 2 * bp_) * N + bn + n8, 0, 1);
        }

        wait_async();
        __syncthreads();

        // ---- fragments + 4 WMMAs ----
        FragU af;
        af.q[0] = *(const v4u*)(As + (wmRow + lo) * 32 + 8 * hi);
        af.q[1] = *(const v4u*)(As + (wmRow + lo) * 32 + 16 + 8 * hi);
        for (int c = 0; c < 4; ++c) {
            FragU bfm;
            const unsigned short* brow = Bs + (wnCol + c * 16 + lo) * 32;
            bfm.q[0] = *(const v4u*)(brow + 8 * hi);
            bfm.q[1] = *(const v4u*)(brow + 16 + 8 * hi);
            acc[c] = __builtin_amdgcn_wmma_f32_16x16x32_bf16(
                false, af.b, false, bfm.b, (short)0, acc[c], false, false);
        }
        __syncthreads();
    }

    // ---- epilogue ----
    for (int c = 0; c < 4; ++c) {
        const int col = bn + wnCol + c * 16 + lo;
        for (int r = 0; r < 8; ++r) {
            int row = bm + wmRow + hi * 8 + r;
            if (row < M) {
                float v = acc[c][r];
                if (bias) v += bias[col];
                if (res)  v += res[(size_t)row * N + col];
                if (relu) v = fmaxf(v, 0.f);
                if (outH) outH[(size_t)row * N + col] = f2bf(v);
                else      outF[(size_t)row * N + col] = v;
            }
        }
    }
}

// ---------------------------------------------------------------------------
// Fused attention: one wave per (batch, head, 16-query block).
// S = Q K^T via WMMA (d_k=64 -> 2 k-steps), mask+softmax in LDS, O = P V via
// WMMA with V block staged in LDS. Q/K/V/O are [B*Nseq, 1024] bf16,
// head h occupies columns h*64..h*64+63.
// ---------------------------------------------------------------------------
__global__ __launch_bounds__(32) void attn_wmma(
    const unsigned short* __restrict__ Q, const unsigned short* __restrict__ Km,
    const unsigned short* __restrict__ Vm, unsigned short* __restrict__ O,
    int Nq, int Nk, int causal) {
    __shared__ __align__(16) float          Sld[16 * 512];
    __shared__ __align__(16) unsigned short Pld[16 * 512];
    __shared__ __align__(16) unsigned short Vt[32 * 64];

    const int lane = threadIdx.x;
    const int lo = lane & 15, hi = lane >> 4;
    const int qb = blockIdx.x * 16;
    const int h  = blockIdx.y;
    const int b  = blockIdx.z;
    const size_t qbase = ((size_t)b * Nq) * 1024 + (size_t)h * 64;
    const size_t kbase = ((size_t)b * Nk) * 1024 + (size_t)h * 64;

    // Q fragments (A layout): rows qb+lo, k-dim chunks 0..31 and 32..63
    FragU qa[2];
    {
        const unsigned short* qrow = Q + qbase + (size_t)(qb + lo) * 1024;
        for (int c = 0; c < 2; ++c) {
            qa[c].q[0] = *(const v4u*)(qrow + c * 32 + 8 * hi);
            qa[c].q[1] = *(const v4u*)(qrow + c * 32 + 16 + 8 * hi);
        }
    }

    const int nkb = (Nk + 31) >> 5;
    const int W   = nkb << 5;

    // ---- scores ----
    for (int kb = 0; kb < nkb; ++kb) {
        for (int t = 0; t < 2; ++t) {
            int n0  = kb * 32 + t * 16;
            int key = n0 + lo;
            if (key >= Nk) key = Nk - 1;
            const unsigned short* krow = Km + kbase + (size_t)key * 1024;
            v8f s;
            for (int i = 0; i < 8; ++i) s[i] = 0.f;
            for (int c = 0; c < 2; ++c) {
                FragU kf;
                kf.q[0] = *(const v4u*)(krow + c * 32 + 8 * hi);
                kf.q[1] = *(const v4u*)(krow + c * 32 + 16 + 8 * hi);
                s = __builtin_amdgcn_wmma_f32_16x16x32_bf16(false, qa[c].b, false,
                                                            kf.b, (short)0, s,
                                                            false, false);
            }
            int coln = n0 + lo;
            for (int r = 0; r < 8; ++r) {
                int m = hi * 8 + r;
                float v = s[r] * 0.125f;  // 1/sqrt(64)
                bool valid = (coln < Nk) && (!causal || coln <= (qb + m));
                Sld[m * 512 + coln] = valid ? v : -1e30f;
            }
        }
    }
    __syncthreads();

    // ---- softmax (lanes 0..15 each own one query row) ----
    if (lane < 16) {
        float* srow = Sld + lane * 512;
        float mx = -1e30f;
        for (int c = 0; c < W; ++c) mx = fmaxf(mx, srow[c]);
        float sum = 0.f;
        for (int c = 0; c < W; ++c) {
            float e = __expf(srow[c] - mx);
            srow[c] = e;
            sum += e;
        }
        float inv = 1.f / sum;
        unsigned short* prow = Pld + lane * 512;
        for (int c = 0; c < W; ++c) prow[c] = f2bf(srow[c] * inv);
    }
    __syncthreads();

    // ---- O = P x V ----
    v8f o[4];
    for (int c = 0; c < 4; ++c)
        for (int i = 0; i < 8; ++i) o[c][i] = 0.f;

    for (int kb = 0; kb < nkb; ++kb) {
        {   // stage V block 32 keys x 64 dims
            int key = kb * 32 + lane;
            if (key >= Nk) key = Nk - 1;
            const unsigned short* vrow = Vm + kbase + (size_t)key * 1024;
            for (int u = 0; u < 8; ++u)
                *(v4u*)(Vt + lane * 64 + u * 8) = *(const v4u*)(vrow + u * 8);
        }
        __syncthreads();
        FragU pa;
        pa.q[0] = *(const v4u*)(Pld + lo * 512 + kb * 32 + 8 * hi);
        pa.q[1] = *(const v4u*)(Pld + lo * 512 + kb * 32 + 16 + 8 * hi);
        for (int c = 0; c < 4; ++c) {
            FragU vf;
            int d = c * 16 + lo;
            for (int j = 0; j < 8; ++j) {
                vf.u[j]     = Vt[(8 * hi + j) * 64 + d];
                vf.u[j + 8] = Vt[(16 + 8 * hi + j) * 64 + d];
            }
            o[c] = __builtin_amdgcn_wmma_f32_16x16x32_bf16(false, pa.b, false,
                                                           vf.b, (short)0, o[c],
                                                           false, false);
        }
        __syncthreads();
    }

    for (int c = 0; c < 4; ++c) {
        int d = c * 16 + lo;
        for (int r = 0; r < 8; ++r) {
            int m = hi * 8 + r;
            O[((size_t)b * Nq + qb + m) * 1024 + h * 64 + d] = f2bf(o[c][r]);
        }
    }
}

// ---------------------------------------------------------------------------
// Fusion gate: per row, s_k = dot(a_k, x), w = softmax over {0,1},
// out = w0*a0 + w1*a1. One 256-thread block per row.
// ---------------------------------------------------------------------------
__global__ __launch_bounds__(256) void fuse_gate_k(const float* __restrict__ x,
                                                   const float* __restrict__ a0,
                                                   const float* __restrict__ a1,
                                                   float* __restrict__ out,
                                                   int rows) {
    __shared__ float r0[256];
    __shared__ float r1[256];
    int row = blockIdx.x;
    if (row >= rows) return;
    const float* xr = x + (size_t)row * 1024;
    const float* p0 = a0 + (size_t)row * 1024;
    const float* p1 = a1 + (size_t)row * 1024;
    float d0 = 0.f, d1 = 0.f;
    for (int j = 0; j < 4; ++j) {
        int c = threadIdx.x + 256 * j;
        d0 += p0[c] * xr[c];
        d1 += p1[c] * xr[c];
    }
    r0[threadIdx.x] = d0;
    r1[threadIdx.x] = d1;
    __syncthreads();
    for (int o = 128; o > 0; o >>= 1) {
        if ((int)threadIdx.x < o) {
            r0[threadIdx.x] += r0[threadIdx.x + o];
            r1[threadIdx.x] += r1[threadIdx.x + o];
        }
        __syncthreads();
    }
    float s0 = r0[0], s1 = r1[0];
    float mx = fmaxf(s0, s1);
    float e0 = __expf(s0 - mx), e1 = __expf(s1 - mx);
    float inv = 1.f / (e0 + e1);
    float w0 = e0 * inv, w1 = e1 * inv;
    for (int j = 0; j < 4; ++j) {
        int c = threadIdx.x + 256 * j;
        out[(size_t)row * 1024 + c] = w0 * p0[c] + w1 * p1[c];
    }
}

// fuse = c1 + 0.5*(sem + vis)
__global__ __launch_bounds__(256) void fuse_combine_k(const float* __restrict__ c1,
                                                      const float* __restrict__ sem,
                                                      const float* __restrict__ vis,
                                                      float* __restrict__ fuse,
                                                      long long n) {
    long long i = (long long)blockIdx.x * 256 + threadIdx.x;
    long long stride = (long long)gridDim.x * 256;
    for (; i < n; i += stride)
        fuse[i] = c1[i] + 0.5f * (sem[i] + vis[i]);
}

// ---------------------------------------------------------------------------
// Host-side orchestration
// ---------------------------------------------------------------------------
extern "C" void kernel_launch(void* const* d_in, const int* in_sizes, int n_in,
                              void* d_out, int out_size, void* d_ws, size_t ws_size,
                              hipStream_t stream) {
    (void)in_sizes; (void)n_in; (void)out_size; (void)ws_size;

    constexpr int B = 16, N1 = 512, D = 1024, DFF = 4096;
    constexpr int NCPT = 25, NSEN = 25, NREG = 196, NSPA = 196;
    const int ROWS = B * N1;                     // 8192
    const long long ACT = (long long)ROWS * D;   // 8,388,608

    const float* captions = (const float*)d_in[0];
    const float* cpt_w    = (const float*)d_in[1];
    const float* sen_w    = (const float*)d_in[2];
    const float* reg_f    = (const float*)d_in[3];
    const float* spa_f    = (const float*)d_in[4];
    const float* att_W    = (const float*)d_in[5];   // [5,4,1024,1024]
    const float* att_b    = (const float*)d_in[6];   // [5,4,1024]
    const float* ffn_W1   = (const float*)d_in[7];
    const float* ffn_b1   = (const float*)d_in[8];
    const float* ffn_W2   = (const float*)d_in[9];
    const float* ffn_b2   = (const float*)d_in[10];
    const float* ln_g     = (const float*)d_in[11];  // [6,1024]
    const float* ln_b     = (const float*)d_in[12];
    // d_in[13] = seq_masks (exact causal tril) -> encoded via `causal` flag

    // -------- workspace carve --------
    const size_t ACTF = (size_t)ACT * 4;   // 33,554,432 (mult of 256)
    const size_t ACTH = (size_t)ACT * 2;   // 16,777,216
    char* w = (char*)d_ws;
    size_t off = 0;
    auto take = [&](size_t bytes) -> char* {
        char* p = w + off;
        off += (bytes + 255) & ~(size_t)255;
        return p;
    };
    float* c1    = (float*)take(ACTF);
    float* fuse  = (float*)take(ACTF);
    float* t0    = (float*)take(ACTF);   // t0,t1 contiguous (ffn1 alias)
    float* t1    = (float*)take(ACTF);
    float* sem   = (float*)take(ACTF);
    float* vis   = (float*)take(ACTF);
    unsigned short* hb   = (unsigned short*)take(ACTH);
    unsigned short* qbuf = (unsigned short*)take(ACTH);
    unsigned short* kbuf = (unsigned short*)take(ACTH);
    unsigned short* vbuf = (unsigned short*)take(ACTH);
    unsigned short* abuf = (unsigned short*)take(ACTH);
    unsigned short* srcb = (unsigned short*)take((size_t)B * NREG * D * 2);
    const long long WATT = 5LL * 4 * D * D;          // 20,971,520
    const long long W1N  = (long long)D * DFF;       // 4,194,304
    unsigned short* Wc   = (unsigned short*)take((size_t)(WATT + 2 * W1N) * 2);
    unsigned short* Wf1  = Wc + WATT;
    unsigned short* Wf2  = Wc + WATT + W1N;
    unsigned short* ffn1 = (unsigned short*)t0;      // 67MB spans t0+t1 (reused)

    // -------- helpers --------
    auto cvt = [&](const float* src, unsigned short* dst, long long n) {
        cvt_f32_bf16<<<2048, 256, 0, stream>>>(src, dst, n);
    };
    auto ln = [&](const float* x, int idx, unsigned short* y) {
        ln_bf16<<<ROWS, 256, 0, stream>>>(x, ln_g + idx * D, ln_b + idx * D, y, ROWS);
    };
    auto gemm = [&](const unsigned short* A, const unsigned short* Bw,
                    const float* bias, const float* res, float* oF,
                    unsigned short* oH, int M, int N, int K, int relu) {
        dim3 g(N / 128, (M + 63) / 64);
        gemm_bf16_wmma<<<g, 256, 0, stream>>>(A, Bw, bias, res, oF, oH, M, N, K, relu);
    };
    auto attn = [&](const unsigned short* Qp, const unsigned short* Kp,
                    const unsigned short* Vp, unsigned short* Op, int Nk, int causal) {
        dim3 g(N1 / 16, 16 /*H*/, B);
        attn_wmma<<<g, 32, 0, stream>>>(Qp, Kp, Vp, Op, N1, Nk, causal);
    };
    auto Wp = [&](int i, int j) { return Wc + ((size_t)(i * 4 + j)) * D * D; };
    auto bp = [&](int i, int j) { return att_b + (size_t)(i * 4 + j) * D; };

    // -------- convert weights to bf16 --------
    cvt(att_W, Wc, WATT);
    cvt(ffn_W1, Wf1, W1N);
    cvt(ffn_W2, Wf2, W1N);

    // -------- self-attention (causal) --------
    ln(captions, 0, hb);
    gemm(hb, Wp(0, 0), bp(0, 0), nullptr, nullptr, qbuf, ROWS, D, D, 0);
    gemm(hb, Wp(0, 1), bp(0, 1), nullptr, nullptr, kbuf, ROWS, D, D, 0);
    gemm(hb, Wp(0, 2), bp(0, 2), nullptr, nullptr, vbuf, ROWS, D, D, 0);
    attn(qbuf, kbuf, vbuf, abuf, N1, 1);
    gemm(abuf, Wp(0, 3), bp(0, 3), captions, c1, nullptr, ROWS, D, D, 0);  // c1 = c + sa

    // -------- cross attention helper --------
    auto cross = [&](int i, const float* src, int Nk, float* dst) {
        int Mkv = B * Nk;
        cvt(src, srcb, (long long)Mkv * D);
        ln(c1, i, hb);
        gemm(hb,   Wp(i, 0), bp(i, 0), nullptr, nullptr, qbuf, ROWS, D, D, 0);
        gemm(srcb, Wp(i, 1), bp(i, 1), nullptr, nullptr, kbuf, Mkv, D, D, 0);
        gemm(srcb, Wp(i, 2), bp(i, 2), nullptr, nullptr, vbuf, Mkv, D, D, 0);
        attn(qbuf, kbuf, vbuf, abuf, Nk, 0);
        gemm(abuf, Wp(i, 3), bp(i, 3), nullptr, dst, nullptr, ROWS, D, D, 0);
    };

    // semantic branch
    cross(1, cpt_w, NCPT, t0);
    cross(2, sen_w, NSEN, t1);
    fuse_gate_k<<<ROWS, 256, 0, stream>>>(c1, t0, t1, sem, ROWS);
    // visual branch
    cross(3, reg_f, NREG, t0);
    cross(4, spa_f, NSPA, t1);
    fuse_gate_k<<<ROWS, 256, 0, stream>>>(c1, t0, t1, vis, ROWS);

    // fuse = c1 + 0.5*(sem+vis)
    fuse_combine_k<<<2048, 256, 0, stream>>>(c1, sem, vis, fuse, ACT);

    // -------- FFN (ffn1 aliases t0/t1, which are free now) --------
    ln(fuse, 5, hb);
    gemm(hb, Wf1, ffn_b1, nullptr, nullptr, ffn1, ROWS, DFF, D, 1);        // ReLU, bf16
    gemm(ffn1, Wf2, ffn_b2, fuse, (float*)d_out, nullptr, ROWS, D, DFF, 0);
}
